// SwitchMOE_37503654429110
// MI455X (gfx1250) — matrix operand order repaired and checked by reading, not compiled, not run on hardware
//
#include <hip/hip_runtime.h>
#include <hip/hip_bf16.h>
#include <math.h>

// ---------------------------------------------------------------------------
// Switch-MoE top-1 forward for gfx1250 (MI455X).
//   x[2,2048,1024] f32, gate_w[8,1024], w1[8,4096,1024], b1[8,4096],
//   w2[8,1024,4096], b2[8,1024]  ->  out[2,2048,1024] f32 ++ aux_loss (1 f32)
// Strategy: bf16 weights in workspace + v_wmma_f32_16x16x32_bf16, fp32 accum.
// 2-way N-tile register blocking: one A fragment feeds two WMMAs.
// ---------------------------------------------------------------------------

typedef __attribute__((ext_vector_type(16))) __bf16       v16bf;
typedef __attribute__((ext_vector_type(4)))  __bf16       bf16x4;
typedef __attribute__((ext_vector_type(8)))  float        v8f;
typedef __attribute__((ext_vector_type(4)))  float        f32x4;
typedef __attribute__((ext_vector_type(4)))  unsigned int u32x4;

#define D_MODEL 1024
#define HIDDEN  4096
#define NEXP    8
#define TOK     4096          // B*N
#define LBC     0.01f

// workspace layout (assumes ws_size >= ~136 MB)
#define OFF_CNT   0u                                   // int[NEXP]
#define OFF_PSUM  64u                                  // float[NEXP]
#define OFF_GATE  128u                                 // float[TOK]
#define OFF_LIST  (128u + TOK * 4u)                    // int[NEXP*TOK]
#define OFF_W1B   (1u << 20)                           // bf16[NEXP*HIDDEN*D_MODEL]
#define OFF_W2B   (OFF_W1B + 2u * NEXP * HIDDEN * D_MODEL)

#define XSTR (D_MODEL + 8)    // padded LDS row stride (bf16 elems), 16B aligned
#define HSTR (HIDDEN + 8)

// ---------------------------------------------------------------------------
__global__ void moe_init_kernel(int* cnt, float* psum) {
  int i = threadIdx.x;
  if (i < NEXP) { cnt[i] = 0; psum[i] = 0.0f; }
}

// fp32 -> bf16 conversion of both expert weight tensors (vectorized: b128 in,
// b64 out)
__global__ __launch_bounds__(256) void moe_cvt_kernel(
    const float* __restrict__ w1, const float* __restrict__ w2,
    __bf16* __restrict__ w1b, __bf16* __restrict__ w2b, int n4) {
  const f32x4* s1 = (const f32x4*)w1;
  const f32x4* s2 = (const f32x4*)w2;
  bf16x4* d1 = (bf16x4*)w1b;
  bf16x4* d2 = (bf16x4*)w2b;
  for (int i = blockIdx.x * 256 + threadIdx.x; i < n4; i += gridDim.x * 256) {
    f32x4 a = s1[i], b = s2[i];
    bf16x4 ra, rb;
#pragma unroll
    for (int k = 0; k < 4; ++k) { ra[k] = (__bf16)a[k]; rb[k] = (__bf16)b[k]; }
    d1[i] = ra;
    d2[i] = rb;
  }
}

// router: logits -> softmax -> top-1 bucket + gate prob + aux partial sums
__global__ __launch_bounds__(256) void moe_router_kernel(
    const float* __restrict__ x, const float* __restrict__ gw,
    int* __restrict__ cnt, float* __restrict__ psum,
    int* __restrict__ list, float* __restrict__ gateArr) {
  __shared__ float sgw[NEXP * D_MODEL];   // 32 KB
  __shared__ float sP[NEXP];
  const int tid = threadIdx.x;
  for (int i = tid; i < NEXP * D_MODEL; i += 256) sgw[i] = gw[i];
  if (tid < NEXP) sP[tid] = 0.0f;
  __syncthreads();

  const int t = blockIdx.x * 256 + tid;
  const float* xr = x + (size_t)t * D_MODEL;
  float acc[NEXP] = {};
  for (int d = 0; d < D_MODEL; ++d) {
    float xv = xr[d];
#pragma unroll
    for (int e = 0; e < NEXP; ++e) acc[e] += xv * sgw[e * D_MODEL + d];
  }
  float m = acc[0]; int sel = 0;
#pragma unroll
  for (int e = 1; e < NEXP; ++e) if (acc[e] > m) { m = acc[e]; sel = e; }
  float p[NEXP]; float s = 0.0f;
#pragma unroll
  for (int e = 0; e < NEXP; ++e) { p[e] = __expf(acc[e] - m); s += p[e]; }
  const float inv = 1.0f / s;        // top-1 prob = exp(0)/s
#pragma unroll
  for (int e = 0; e < NEXP; ++e) atomicAdd(&sP[e], p[e] * inv);
  gateArr[t] = inv;
  int pos = atomicAdd(&cnt[sel], 1);
  list[sel * TOK + pos] = t;
  __syncthreads();
  if (tid < NEXP) atomicAdd(&psum[tid], sP[tid]);
}

// fused expert FFN: one block = (expert e, tile of 16 tokens), 8 waves
__global__ __launch_bounds__(256) void moe_ffn_kernel(
    const float* __restrict__ x,
    const float* __restrict__ b1, const float* __restrict__ b2,
    const __bf16* __restrict__ w1b, const __bf16* __restrict__ w2b,
    const int* __restrict__ cnt, const int* __restrict__ list,
    const float* __restrict__ gate, float* __restrict__ out) {
  const int e    = blockIdx.x >> 8;
  const int tile = blockIdx.x & 255;
  const int cntE = cnt[e];
  const int base = tile * 16;
  if (base >= cntE) return;

  extern __shared__ char smem[];
  __bf16* lx = (__bf16*)smem;                                    // 16*XSTR bf16
  __bf16* lh = (__bf16*)(smem + 16 * XSTR * 2);                  // 16*HSTR bf16
  int*   stok  = (int*)(smem + 16 * XSTR * 2 + 16 * HSTR * 2);
  float* sgate = (float*)(smem + 16 * XSTR * 2 + 16 * HSTR * 2 + 64);

  const int tid  = threadIdx.x;
  const int lane = tid & 31;
  const int wave = tid >> 5;
  const int nn   = lane & 15;      // N column (and A row index)
  const int hi   = lane >> 4;      // lane half-group

  const int* listE = list + e * TOK;
  if (tid < 16) {
    int idx = base + tid;
    int cl  = idx < cntE ? idx : (cntE - 1);
    int t   = listE[cl];
    stok[tid]  = t;
    sgate[tid] = gate[t];
  }
  __syncthreads();

  // stage 16 gathered x rows into LDS as bf16 (vectorized b128->b64)
  for (int i4 = tid; i4 < 16 * (D_MODEL / 4); i4 += 256) {
    int r = i4 >> 8, c = (i4 & 255) << 2;
    f32x4 v = *(const f32x4*)(x + (size_t)stok[r] * D_MODEL + c);
    bf16x4 o;
#pragma unroll
    for (int k = 0; k < 4; ++k) o[k] = (__bf16)v[k];
    *(bf16x4*)(lx + r * XSTR + c) = o;
  }
  __syncthreads();

  const __bf16* w1e = w1b + (size_t)e * HIDDEN * D_MODEL;
  const __bf16* w2e = w2b + (size_t)e * D_MODEL * HIDDEN;
  const float*  b1e = b1 + e * HIDDEN;
  const float*  b2e = b2 + e * D_MODEL;

  union AB { u32x4 q[2]; v16bf v; };

  // ---- GEMM1: h = gelu(x @ w1^T + b1). Each wave: 16 iters x 2 N-tiles ----
  for (int j = 0; j < 16; ++j) {
    const int h0 = (wave * 32 + j * 2) << 4;   // first of two adjacent N-tiles
    v8f acc0 = {}, acc1 = {};
    const __bf16* arow  = lx + nn * XSTR + hi * 16;                     // A M=nn
    const __bf16* bcol0 = w1e + (size_t)(h0 + nn) * D_MODEL + hi * 16;  // N-tile0
    const __bf16* bcol1 = bcol0 + (size_t)16 * D_MODEL;                 // N-tile1
#pragma unroll 4
    for (int k0 = 0; k0 < D_MODEL; k0 += 32) {
      AB a, b0, b1f;
      const __bf16* ap = arow + k0;
      a.q[0] = *(const u32x4*)(ap);        // K = k0 + hi*16 .. +7
      a.q[1] = *(const u32x4*)(ap + 8);    // K = k0 + hi*16+8 .. +15
      const __bf16* bp0 = bcol0 + k0;
      b0.q[0] = *(const u32x4*)(bp0);      // K = k0 + hi*16 .. +7
      b0.q[1] = *(const u32x4*)(bp0 + 8);  // K = k0 + hi*16+8 .. +15
      const __bf16* bp1 = bcol1 + k0;
      b1f.q[0] = *(const u32x4*)(bp1);
      b1f.q[1] = *(const u32x4*)(bp1 + 8);
      acc0 = __builtin_amdgcn_wmma_f32_16x16x32_bf16(
          false, a.v, false, b0.v, (short)0, acc0, false, false);
      acc1 = __builtin_amdgcn_wmma_f32_16x16x32_bf16(
          false, a.v, false, b1f.v, (short)0, acc1, false, false);
    }
    const float bias0 = b1e[h0 + nn];
    const float bias1 = b1e[h0 + 16 + nn];
#pragma unroll
    for (int r = 0; r < 8; ++r) {
      const int M = r + hi * 8;
      float v0 = acc0[r] + bias0;
      float v1 = acc1[r] + bias1;
      float g0 = 0.5f * v0 * (1.0f + erff(v0 * 0.70710678f));  // exact gelu
      float g1 = 0.5f * v1 * (1.0f + erff(v1 * 0.70710678f));
      lh[M * HSTR + h0 + nn]      = (__bf16)g0;
      lh[M * HSTR + h0 + 16 + nn] = (__bf16)g1;
    }
  }
  __syncthreads();

  // ---- GEMM2: y = h @ w2^T + b2; out[tok] = gate * y. 4 iters x 2 tiles ---
  for (int j = 0; j < 4; ++j) {
    const int d0 = (wave * 8 + j * 2) << 4;
    v8f acc0 = {}, acc1 = {};
    const __bf16* arow  = lh + nn * HSTR + hi * 16;
    const __bf16* bcol0 = w2e + (size_t)(d0 + nn) * HIDDEN + hi * 16;
    const __bf16* bcol1 = bcol0 + (size_t)16 * HIDDEN;
#pragma unroll 4
    for (int k0 = 0; k0 < HIDDEN; k0 += 32) {
      AB a, b0, b1f;
      const __bf16* ap = arow + k0;
      a.q[0] = *(const u32x4*)(ap);
      a.q[1] = *(const u32x4*)(ap + 8);
      const __bf16* bp0 = bcol0 + k0;
      b0.q[0] = *(const u32x4*)(bp0);
      b0.q[1] = *(const u32x4*)(bp0 + 8);
      const __bf16* bp1 = bcol1 + k0;
      b1f.q[0] = *(const u32x4*)(bp1);
      b1f.q[1] = *(const u32x4*)(bp1 + 8);
      acc0 = __builtin_amdgcn_wmma_f32_16x16x32_bf16(
          false, a.v, false, b0.v, (short)0, acc0, false, false);
      acc1 = __builtin_amdgcn_wmma_f32_16x16x32_bf16(
          false, a.v, false, b1f.v, (short)0, acc1, false, false);
    }
    const float bias0 = b2e[d0 + nn];
    const float bias1 = b2e[d0 + 16 + nn];
#pragma unroll
    for (int r = 0; r < 8; ++r) {
      const int M = r + hi * 8;
      if (base + M < cntE) {
        const size_t rowoff = (size_t)stok[M] * D_MODEL;
        out[rowoff + d0 + nn]      = (acc0[r] + bias0) * sgate[M];
        out[rowoff + d0 + 16 + nn] = (acc1[r] + bias1) * sgate[M];
      }
    }
  }
}

__global__ void moe_aux_kernel(const int* cnt, const float* psum,
                               float* out, int idx) {
  float loss = 0.0f;
  for (int e = 0; e < NEXP; ++e)
    loss += ((float)cnt[e] / (float)TOK) * (psum[e] / (float)TOK);
  out[idx] = LBC * (float)NEXP * loss;
}

// ---------------------------------------------------------------------------
extern "C" void kernel_launch(void* const* d_in, const int* in_sizes, int n_in,
                              void* d_out, int out_size, void* d_ws, size_t ws_size,
                              hipStream_t stream) {
  (void)in_sizes; (void)n_in; (void)ws_size;
  const float* x  = (const float*)d_in[0];
  const float* gw = (const float*)d_in[1];
  const float* w1 = (const float*)d_in[2];
  const float* b1 = (const float*)d_in[3];
  const float* w2 = (const float*)d_in[4];
  const float* b2 = (const float*)d_in[5];
  float* out = (float*)d_out;
  char*  ws  = (char*)d_ws;

  int*    cnt  = (int*)(ws + OFF_CNT);
  float*  psum = (float*)(ws + OFF_PSUM);
  float*  gate = (float*)(ws + OFF_GATE);
  int*    list = (int*)(ws + OFF_LIST);
  __bf16* w1b  = (__bf16*)(ws + OFF_W1B);
  __bf16* w2b  = (__bf16*)(ws + OFF_W2B);

  moe_init_kernel<<<1, 32, 0, stream>>>(cnt, psum);
  moe_cvt_kernel<<<8192, 256, 0, stream>>>(w1, w2, w1b, w2b,
                                           NEXP * HIDDEN * D_MODEL / 4);
  moe_router_kernel<<<TOK / 256, 256, 0, stream>>>(x, gw, cnt, psum, list, gate);

  const size_t lds = (size_t)16 * XSTR * 2 + (size_t)16 * HSTR * 2 + 128;
  moe_ffn_kernel<<<NEXP * 256, 256, lds, stream>>>(x, b1, b2, w1b, w2b,
                                                   cnt, list, gate, out);
  moe_aux_kernel<<<1, 1, 0, stream>>>(cnt, psum, out, out_size - 1);
}
